// HGNNLayer_31894427140524
// MI455X (gfx1250) — matrix operand order repaired and checked by reading, not compiled, not run on hardware
//
#include <hip/hip_runtime.h>
#include <hip/hip_bf16.h>

// ---------------------------------------------------------------------------
// HGNN layer for MI455X (gfx1250, wave32, WMMA + TDM).
// - All dense GEMMs: v_wmma_f32_16x16x32_bf16 (bf16 in, f32 accumulate).
// - Weights are converted+transposed to bf16 ONCE per launch; each GEMM block
//   then DMAs its [ncols x 32] K-panels into LDS with TENSOR_LOAD_TO_LDS
//   (TENSORcnt + s_wait_tensorcnt + workgroup barrier), removing the fp32
//   weight traffic and f32->bf16 VALU conversion from the GEMM inner loop.
// - Sparse gathers read bf16 activation mirrors; scatters use fp32 atomics.
// ---------------------------------------------------------------------------

typedef __attribute__((ext_vector_type(16))) __bf16 v16bf;
typedef __attribute__((ext_vector_type(8)))  float  v8f;
typedef __attribute__((ext_vector_type(4)))  unsigned v4u;
typedef __attribute__((ext_vector_type(8)))  int      v8i;
typedef __attribute__((ext_vector_type(4)))  int      v4i;
typedef unsigned short u16;

#define DD 128   // feature dim

#if __has_builtin(__builtin_amdgcn_tensor_load_to_lds)
#define HGNN_TDM 1
#endif

__device__ inline u16 f2bf(float f) {
  unsigned u = __float_as_uint(f);
  unsigned r = (u + 0x7FFFu + ((u >> 16) & 1u)) >> 16;   // round-to-nearest-even
  return (u16)r;
}
__device__ inline float bf2f(u16 h) { return __uint_as_float(((unsigned)h) << 16); }

// order-preserving float <-> uint encoding for atomicMax on floats
__device__ inline unsigned fenc(float f) {
  unsigned u = __float_as_uint(f);
  return (u & 0x80000000u) ? ~u : (u | 0x80000000u);
}
__device__ inline float fdec(unsigned u) {
  return __uint_as_float((u & 0x80000000u) ? (u & 0x7FFFFFFFu) : ~u);
}

__device__ inline v8f v8zero() {
  v8f z;
  for (int j = 0; j < 8; ++j) z[j] = 0.f;
  return z;
}

union FragU { unsigned u[8]; v16bf v; };

// A fragment 16x32 bf16 from LDS (row-major, `stride` elements per row).
// ISA 7.12.2: lanes 0-15 -> M=lane, K in {0..7,16..23}; lanes 16-31 -> M=lane-16,
// K in {8..15,24..31}; two packed bf16 per VGPR.
__device__ inline v16bf frag_a(const u16* base, int stride, int row0, int k0) {
  const int lane = threadIdx.x & 31;
  const u16* p = base + (row0 + (lane & 15)) * stride;
  const int kb = k0 + ((lane >> 4) << 3);
  FragU f;
#pragma unroll
  for (int j = 0; j < 8; ++j) {
    int k = kb + ((j & 4) << 2) + ((j & 3) << 1);
    f.u[j] = *(const unsigned*)(p + k);
  }
  return f.v;
}

// B fragment 32x16 bf16 from LDS stored N-major (Wt[n*stride + k]).
// Same K pattern as frag_a: lane 0-15 -> N=lane, K in {0..7,16..23};
// lane 16-31 -> N=lane-16, K in {8..15,24..31}; two packed bf16 per VGPR.
__device__ inline v16bf frag_b(const u16* base, int stride, int n0, int k0) {
  const int lane = threadIdx.x & 31;
  const u16* p = base + (n0 + (lane & 15)) * stride;
  const int kb = k0 + ((lane >> 4) << 3);
  FragU f;
#pragma unroll
  for (int j = 0; j < 8; ++j) f.u[j] = *(const unsigned*)(p + kb + ((j & 4) << 2) + ((j & 3) << 1));
  return f.v;
}

__device__ inline v8f wmma_bf16(v16bf a, v16bf b, v8f c) {
  return __builtin_amdgcn_wmma_f32_16x16x32_bf16(false, a, false, b, (short)0, c,
                                                 false, false);
}

// ---------------------------------------------------------------------------
// Tensor Data Mover: DMA a 2D tile (tile_d1 rows x tile_d0 elems) of bf16 from
// global to LDS. D# per ISA ch.8: group0 = count/lds_addr/global_addr/type,
// group1 = data_size, tensor dims, tile dims, dim0 stride. Issued by one wave
// (EXEC is ignored by TDM); completion via s_wait_tensorcnt then barrier.
// ---------------------------------------------------------------------------
__device__ inline void tdm_tile_load_bf16(const void* gsrc, void* ldst,
                                          unsigned td0, unsigned td1,
                                          unsigned tl0, unsigned tl1,
                                          unsigned st0) {
#ifdef HGNN_TDM
  unsigned lds_addr = (unsigned)(unsigned long long)(size_t)ldst;  // low 32b = LDS offset
  unsigned long long ga = (unsigned long long)(size_t)gsrc;
  v4u g0 = { 1u,                                   // count=1 (valid user D#)
             lds_addr,
             (unsigned)ga,                         // global_addr[31:0]
             ((unsigned)(ga >> 32) & 0x01FFFFFFu) | 0x80000000u };  // addr[56:32] | type=2
  v8i g1 = { (int)(1u << 16),                      // data_size=1 -> 2 bytes
             (int)(td0 << 16),                     // tensor_dim0[15:0]
             (int)((td0 >> 16) | (td1 << 16)),     // tensor_dim0[31:16] | tensor_dim1[15:0]
             (int)((td1 >> 16) | (tl0 << 16)),     // tensor_dim1[31:16] | tile_dim0
             (int)(tl1 & 0xFFFFu),                 // tile_dim1 | tile_dim2=0
             (int)st0,                             // tensor_dim0_stride[31:0]
             0, 0 };
  v4i gz = {0, 0, 0, 0};
#if defined(__clang_major__) && (__clang_major__ >= 23)
  v8i gz8 = {0, 0, 0, 0, 0, 0, 0, 0};
  __builtin_amdgcn_tensor_load_to_lds(g0, g1, gz, gz, gz8, 0);
#else
  __builtin_amdgcn_tensor_load_to_lds(g0, g1, gz, gz, 0);
#endif
#endif
}

// Stage K-panel kp of an N-major bf16 weight Wt[ncols][K] into sWp[ncols][32].
// Ends with a workgroup barrier that publishes the LDS tile.
__device__ inline void stage_wpanel(const u16* __restrict__ Wt, int K, int ncols,
                                    int kp, u16* sWp) {
#ifdef HGNN_TDM
  if (threadIdx.x < 32) {   // one wave issues the DMA
    tdm_tile_load_bf16(Wt + (size_t)kp * 32, sWp,
                       (unsigned)K, (unsigned)ncols,   // tensor dims
                       32u, (unsigned)ncols,           // tile dims
                       (unsigned)K);                   // dim0 stride
#if __has_builtin(__builtin_amdgcn_s_wait_tensorcnt)
    __builtin_amdgcn_s_wait_tensorcnt(0);
#endif
  }
#else
  for (int i = threadIdx.x; i < ncols * 8; i += blockDim.x) {  // 8 uint2 per 32-bf16 row
    int n = i >> 3, c = i & 7;
    ((uint2*)(sWp + n * 32))[c] = ((const uint2*)(Wt + (size_t)n * K + kp * 32))[c];
  }
#endif
  __syncthreads();
}

// one-time per launch: Wt[n*K + k] = bf16(W[k*K + n])  (square KxK weights)
__global__ void transpose_bf16_kernel(const float* __restrict__ W,
                                      u16* __restrict__ Wt, int K, int kshift) {
  const int total = K * K;
  for (int i = blockIdx.x * blockDim.x + threadIdx.x; i < total;
       i += gridDim.x * blockDim.x) {
    int n = i >> kshift, k = i & (K - 1);
    Wt[(size_t)n * K + k] = f2bf(W[(size_t)k * K + n]);
  }
}

// ---------------------------------------------------------------------------
// Fused 2-layer MLP: Y = relu(X@W1+b1)@W2+b2, D=128, 64 rows per block.
// Weights arrive pre-transposed bf16 (N-major); panels staged via TDM.
// ---------------------------------------------------------------------------
__global__ __launch_bounds__(256) void dense_mlp_kernel(
    const float* __restrict__ X,
    const u16* __restrict__ W1t, const float* __restrict__ b1,
    const u16* __restrict__ W2t, const float* __restrict__ b2,
    float* __restrict__ Y, u16* __restrict__ Ybf, int rows) {
  __shared__ __align__(16) u16 sA[64 * DD];
  __shared__ __align__(16) u16 sT[64 * DD];
  __shared__ __align__(16) u16 sWp[DD * 32];
  __shared__ float sBias[DD];
  const int tid = threadIdx.x, lane = tid & 31, wave = tid >> 5;
  const int row0 = blockIdx.x * 64;

  for (int i = tid; i < 64 * 32; i += 256) {
    int r = i >> 5, c4 = i & 31;
    int gr = row0 + r; if (gr > rows - 1) gr = rows - 1;
    float4 v = ((const float4*)(X + (size_t)gr * DD))[c4];
    u16* d = sA + r * DD + c4 * 4;
    d[0] = f2bf(v.x); d[1] = f2bf(v.y); d[2] = f2bf(v.z); d[3] = f2bf(v.w);
  }
  if (tid < DD) sBias[tid] = b1[tid];
  v8f acc[4];
#pragma unroll
  for (int t = 0; t < 4; ++t) acc[t] = v8zero();
  __syncthreads();

#pragma unroll 1
  for (int kp = 0; kp < 4; ++kp) {
    if (kp < 3) __builtin_prefetch(W1t + (size_t)(kp + 1) * 32, 0, 3);
    stage_wpanel(W1t, DD, DD, kp, sWp);
#pragma unroll
    for (int t = 0; t < 4; ++t) {
      int tile = wave * 4 + t, strip = tile >> 3, col = tile & 7;
      acc[t] = wmma_bf16(frag_a(sA, DD, strip * 16, kp * 32),
                         frag_b(sWp, 32, col * 16, 0), acc[t]);
    }
    __syncthreads();
  }
  // relu(acc + b1) -> sT (bf16)
#pragma unroll
  for (int t = 0; t < 4; ++t) {
    int tile = wave * 4 + t, strip = tile >> 3, col = tile & 7;
    int n = col * 16 + (lane & 15), mb = strip * 16 + ((lane >> 4) << 3);
    float bias = sBias[n];
#pragma unroll
    for (int j = 0; j < 8; ++j) {
      float v = acc[t][j] + bias;
      sT[(mb + j) * DD + n] = f2bf(fmaxf(v, 0.f));
    }
    acc[t] = v8zero();
  }
  __syncthreads();
  if (tid < DD) sBias[tid] = b2[tid];
  __syncthreads();

#pragma unroll 1
  for (int kp = 0; kp < 4; ++kp) {
    stage_wpanel(W2t, DD, DD, kp, sWp);
#pragma unroll
    for (int t = 0; t < 4; ++t) {
      int tile = wave * 4 + t, strip = tile >> 3, col = tile & 7;
      acc[t] = wmma_bf16(frag_a(sT, DD, strip * 16, kp * 32),
                         frag_b(sWp, 32, col * 16, 0), acc[t]);
    }
    __syncthreads();
  }
#pragma unroll
  for (int t = 0; t < 4; ++t) {
    int tile = wave * 4 + t, strip = tile >> 3, col = tile & 7;
    int n = col * 16 + (lane & 15), mb = strip * 16 + ((lane >> 4) << 3);
    float bias = sBias[n];
#pragma unroll
    for (int j = 0; j < 8; ++j) {
      int gm = row0 + mb + j;
      if (gm < rows) {
        float v = acc[t][j] + bias;
        Y[(size_t)gm * DD + n] = v;
        Ybf[(size_t)gm * DD + n] = f2bf(v);
      }
    }
  }
}

// ---------------------------------------------------------------------------
// Fused attention MLP over nnz rows: gather [xA[idxA[i]], xB[idxB[i]]] (256),
// leaky_relu(x@W1+b1) then dot with W2 + b2 -> out[i].  32 rows / block.
// ---------------------------------------------------------------------------
__global__ __launch_bounds__(256) void attn_kernel(
    int nnz, const int* __restrict__ idxA, const int* __restrict__ idxB,
    const u16* __restrict__ XA, const u16* __restrict__ XB,
    const u16* __restrict__ W1t, const float* __restrict__ b1,
    const float* __restrict__ W2, const float* __restrict__ b2,
    float* __restrict__ out) {
  __shared__ __align__(16) u16 sA[32 * 256];
  __shared__ __align__(16) u16 sT[32 * 256];
  __shared__ __align__(16) u16 sWp[256 * 32];
  __shared__ float sB1[256];
  __shared__ float sW2[256];
  const int tid = threadIdx.x, lane = tid & 31, wave = tid >> 5;
  const int base = blockIdx.x * 32;

  { // gather 32 x 256 bf16
    int r = tid >> 3;
    int c0 = (tid & 7) * 32;
    int nz = base + r; if (nz > nnz - 1) nz = nnz - 1;
    const u16* src = (c0 < 128) ? (XA + (size_t)idxA[nz] * 128 + c0)
                                : (XB + (size_t)idxB[nz] * 128 + (c0 - 128));
    const unsigned* s32 = (const unsigned*)src;
    unsigned* d32 = (unsigned*)(sA + r * 256 + c0);
#pragma unroll
    for (int j = 0; j < 16; ++j) d32[j] = s32[j];
  }
  sB1[tid] = b1[tid];
  sW2[tid] = W2[tid];
  v8f acc[4];
#pragma unroll
  for (int t = 0; t < 4; ++t) acc[t] = v8zero();
  __syncthreads();

#pragma unroll 1
  for (int kp = 0; kp < 8; ++kp) {
    if (kp < 7) __builtin_prefetch(W1t + (size_t)(kp + 1) * 32, 0, 3);
    stage_wpanel(W1t, 256, 256, kp, sWp);
#pragma unroll
    for (int t = 0; t < 4; ++t) {
      int tile = wave * 4 + t, strip = tile >> 4, col = tile & 15;
      acc[t] = wmma_bf16(frag_a(sA, 256, strip * 16, kp * 32),
                         frag_b(sWp, 32, col * 16, 0), acc[t]);
    }
    __syncthreads();
  }
  // leaky_relu (slope 0.01) -> sT
#pragma unroll
  for (int t = 0; t < 4; ++t) {
    int tile = wave * 4 + t, strip = tile >> 4, col = tile & 15;
    int n = col * 16 + (lane & 15), mb = strip * 16 + ((lane >> 4) << 3);
    float bias = sB1[n];
#pragma unroll
    for (int j = 0; j < 8; ++j) {
      float v = acc[t][j] + bias;
      v = (v > 0.f) ? v : 0.01f * v;
      sT[(mb + j) * 256 + n] = f2bf(v);
    }
  }
  __syncthreads();
  // a_i = t_i . W2 + b2   (8 threads per row)
  int r = tid >> 3, sub = tid & 7;
  float s = 0.f;
  for (int k = sub; k < 256; k += 8) s += bf2f(sT[r * 256 + k]) * sW2[k];
  s += __shfl_xor(s, 4, 8);
  s += __shfl_xor(s, 2, 8);
  s += __shfl_xor(s, 1, 8);
  int nz = base + r;
  if (sub == 0 && nz < nnz) out[nz] = s + b2[0];
}

// ---------------------------------------------------------------------------
// COO spmm scatter: dst[didx[i]] += val[i] * src[sidx[i]]   (one wave / nnz)
// ---------------------------------------------------------------------------
__global__ void spmm_scatter_kernel(float* __restrict__ dst,
                                    const int* __restrict__ didx,
                                    const int* __restrict__ sidx,
                                    const float* __restrict__ val,
                                    const float* __restrict__ src, int nnz) {
  const int lane = threadIdx.x & 31;
  int gw = (blockIdx.x * blockDim.x + threadIdx.x) >> 5;
  const int nw = (gridDim.x * blockDim.x) >> 5;
  for (int i = gw; i < nnz; i += nw) {
    int d = didx[i], s = sidx[i];
    float v = val[i];
    float4 x = ((const float4*)(src + (size_t)s * DD))[lane];
    float* dp = dst + (size_t)d * DD + lane * 4;
    atomicAdd(dp + 0, v * x.x);
    atomicAdd(dp + 1, v * x.y);
    atomicAdd(dp + 2, v * x.z);
    atomicAdd(dp + 3, v * x.w);
  }
}

// pooled[didx[i]] += w*src[sidx[i]], rs[didx[i]] += w,  w = exp(a[i]-amax)
__global__ void weighted_scatter_kernel(float* __restrict__ pooled,
                                        float* __restrict__ rs,
                                        const int* __restrict__ didx,
                                        const int* __restrict__ sidx,
                                        const float* __restrict__ a,
                                        const unsigned* __restrict__ amax_enc,
                                        const float* __restrict__ src, int nnz) {
  const int lane = threadIdx.x & 31;
  const float amax = fdec(*amax_enc);
  int gw = (blockIdx.x * blockDim.x + threadIdx.x) >> 5;
  const int nw = (gridDim.x * blockDim.x) >> 5;
  for (int i = gw; i < nnz; i += nw) {
    int d = didx[i], s = sidx[i];
    float w = expf(a[i] - amax);
    if (lane == 0) atomicAdd(&rs[d], w);
    float4 x = ((const float4*)(src + (size_t)s * DD))[lane];
    float* dp = pooled + (size_t)d * DD + lane * 4;
    atomicAdd(dp + 0, w * x.x);
    atomicAdd(dp + 1, w * x.y);
    atomicAdd(dp + 2, w * x.z);
    atomicAdd(dp + 3, w * x.w);
  }
}

__global__ void maxreduce_kernel(const float* __restrict__ a, int n,
                                 unsigned* __restrict__ out) {
  __shared__ float sm[256];
  float m = -INFINITY;
  for (int i = blockIdx.x * blockDim.x + threadIdx.x; i < n;
       i += gridDim.x * blockDim.x)
    m = fmaxf(m, a[i]);
  sm[threadIdx.x] = m;
  __syncthreads();
  for (int s = 128; s > 0; s >>= 1) {
    if ((int)threadIdx.x < s) sm[threadIdx.x] = fmaxf(sm[threadIdx.x], sm[threadIdx.x + s]);
    __syncthreads();
  }
  if (threadIdx.x == 0) atomicMax(out, fenc(sm[0]));
}

// buf[r,:] *= scale[r]; write bf16 mirror
__global__ void rowscale_bf16_kernel(float* __restrict__ buf,
                                     const float* __restrict__ scale,
                                     u16* __restrict__ obf, int rows) {
  const int total = rows * DD;
  for (int i = blockIdx.x * blockDim.x + threadIdx.x; i < total;
       i += gridDim.x * blockDim.x) {
    float v = buf[i] * scale[i >> 7];
    buf[i] = v;
    obf[i] = f2bf(v);
  }
}

// dst[r,:] = num[r,:] / (rs[r] + 1e-20)   (dst may alias num)
__global__ void rowdiv_kernel(float* __restrict__ dst, const float* __restrict__ num,
                              const float* __restrict__ rs, int rows) {
  const int total = rows * DD;
  for (int i = blockIdx.x * blockDim.x + threadIdx.x; i < total;
       i += gridDim.x * blockDim.x)
    dst[i] = num[i] / (rs[i >> 7] + 1e-20f);
}

// per-column mean / biased variance; one block per column
__global__ void bn_stats_kernel(const float* __restrict__ x, int rows,
                                float* __restrict__ mu, float* __restrict__ var) {
  const int c = blockIdx.x;
  float s = 0.f, ss = 0.f;
  for (int r = threadIdx.x; r < rows; r += blockDim.x) {
    float v = x[(size_t)r * DD + c];
    s += v;
    ss += v * v;
  }
  __shared__ float as[256], bs[256];
  as[threadIdx.x] = s;
  bs[threadIdx.x] = ss;
  __syncthreads();
  for (int st = 128; st > 0; st >>= 1) {
    if ((int)threadIdx.x < st) {
      as[threadIdx.x] += as[threadIdx.x + st];
      bs[threadIdx.x] += bs[threadIdx.x + st];
    }
    __syncthreads();
  }
  if (threadIdx.x == 0) {
    float m = as[0] / (float)rows;
    mu[c] = m;
    var[c] = bs[0] / (float)rows - m * m;
  }
}

__global__ void bn_tanh_kernel(float* __restrict__ x, const float* __restrict__ mu,
                               const float* __restrict__ var,
                               const float* __restrict__ g,
                               const float* __restrict__ b, int rows) {
  const int total = rows * DD;
  for (int i = blockIdx.x * blockDim.x + threadIdx.x; i < total;
       i += gridDim.x * blockDim.x) {
    int c = i & (DD - 1);
    float v = (x[i] - mu[c]) * rsqrtf(var[c] + 1e-5f) * g[c] + b[c];
    x[i] = tanhf(v);
  }
}

__global__ void init_red_kernel(unsigned* red) {
  if (threadIdx.x == 0) {
    red[0] = fenc(-INFINITY);
    red[1] = fenc(-INFINITY);
  }
}

// ---------------------------------------------------------------------------
// GRU part A: z = sigmoid(h@Wiz+biz + hv@Whz+bhz), r = sigmoid(...)
// outputs Z (f32) and RH = bf16(r * hv)
// ---------------------------------------------------------------------------
__global__ __launch_bounds__(256) void gru_a_kernel(
    const float* __restrict__ H, const float* __restrict__ HV,
    const u16* __restrict__ Wizt, const float* __restrict__ biz,
    const u16* __restrict__ Whzt, const float* __restrict__ bhz,
    const u16* __restrict__ Wirt, const float* __restrict__ bir,
    const u16* __restrict__ Whrt, const float* __restrict__ bhr,
    float* __restrict__ Z, u16* __restrict__ RH, int rows) {
  __shared__ __align__(16) u16 sH[64 * DD];
  __shared__ __align__(16) u16 sV[64 * DD];
  __shared__ __align__(16) u16 sWp[DD * 32];
  const int tid = threadIdx.x, lane = tid & 31, wave = tid >> 5;
  const int row0 = blockIdx.x * 64;
  for (int i = tid; i < 64 * 32; i += 256) {
    int r = i >> 5, c4 = i & 31;
    int gr = row0 + r; if (gr > rows - 1) gr = rows - 1;
    float4 a = ((const float4*)(H + (size_t)gr * DD))[c4];
    float4 b = ((const float4*)(HV + (size_t)gr * DD))[c4];
    u16* dh = sH + r * DD + c4 * 4;
    u16* dv = sV + r * DD + c4 * 4;
    dh[0] = f2bf(a.x); dh[1] = f2bf(a.y); dh[2] = f2bf(a.z); dh[3] = f2bf(a.w);
    dv[0] = f2bf(b.x); dv[1] = f2bf(b.y); dv[2] = f2bf(b.z); dv[3] = f2bf(b.w);
  }
  v8f az[4], ar[4];
#pragma unroll
  for (int t = 0; t < 4; ++t) { az[t] = v8zero(); ar[t] = v8zero(); }
  __syncthreads();

#pragma unroll 1
  for (int kp = 0; kp < 4; ++kp) {
    const u16* Ws[4] = {Wizt, Whzt, Wirt, Whrt};
#pragma unroll 1
    for (int w = 0; w < 4; ++w) {
      stage_wpanel(Ws[w], DD, DD, kp, sWp);
      const u16* src = (w & 1) ? sV : sH;
#pragma unroll
      for (int t = 0; t < 4; ++t) {
        int tile = wave * 4 + t, strip = tile >> 3, col = tile & 7;
        v16bf af = frag_a(src, DD, strip * 16, kp * 32);
        v16bf bf = frag_b(sWp, 32, col * 16, 0);
        if (w < 2) az[t] = wmma_bf16(af, bf, az[t]);
        else       ar[t] = wmma_bf16(af, bf, ar[t]);
      }
      __syncthreads();
    }
  }
#pragma unroll
  for (int t = 0; t < 4; ++t) {
    int tile = wave * 4 + t, strip = tile >> 3, col = tile & 7;
    int n = col * 16 + (lane & 15), mb = strip * 16 + ((lane >> 4) << 3);
    float bz = biz[n] + bhz[n];
    float br = bir[n] + bhr[n];
#pragma unroll
    for (int j = 0; j < 8; ++j) {
      int gm = row0 + mb + j;
      if (gm < rows) {
        float z = 1.f / (1.f + __expf(-(az[t][j] + bz)));
        float r = 1.f / (1.f + __expf(-(ar[t][j] + br)));
        float hv = HV[(size_t)gm * DD + n];
        Z[(size_t)gm * DD + n] = z;
        RH[(size_t)gm * DD + n] = f2bf(r * hv);
      }
    }
  }
}

// GRU part B: n = tanh(h@Win+bin + rh@Whn+bhn); out = (1-z)*n + z*hv
__global__ __launch_bounds__(256) void gru_b_kernel(
    const float* __restrict__ H, const u16* __restrict__ RHg,
    const u16* __restrict__ Wint, const float* __restrict__ bin_,
    const u16* __restrict__ Whnt, const float* __restrict__ bhn,
    const float* __restrict__ Z, const float* __restrict__ HV,
    float* __restrict__ OUT, int rows) {
  __shared__ __align__(16) u16 sH[64 * DD];
  __shared__ __align__(16) u16 sR[64 * DD];
  __shared__ __align__(16) u16 sWp[DD * 32];
  const int tid = threadIdx.x, lane = tid & 31, wave = tid >> 5;
  const int row0 = blockIdx.x * 64;

#ifdef HGNN_TDM
  // rh tile is bf16 and contiguous: 1-D TDM tile; OOB rows read back as zero.
  if (tid < 32) {
    unsigned remain = (unsigned)((rows - row0) < 64 ? (rows - row0) : 64) * DD;
    tdm_tile_load_bf16(RHg + (size_t)row0 * DD, sR,
                       remain, 1u, (unsigned)(64 * DD), 0u, remain);
#if __has_builtin(__builtin_amdgcn_s_wait_tensorcnt)
    __builtin_amdgcn_s_wait_tensorcnt(0);
#endif
  }
#endif
  for (int i = tid; i < 64 * 32; i += 256) {
    int r = i >> 5, c4 = i & 31;
    int gr = row0 + r; if (gr > rows - 1) gr = rows - 1;
    float4 a = ((const float4*)(H + (size_t)gr * DD))[c4];
    u16* dh = sH + r * DD + c4 * 4;
    dh[0] = f2bf(a.x); dh[1] = f2bf(a.y); dh[2] = f2bf(a.z); dh[3] = f2bf(a.w);
#ifndef HGNN_TDM
    *(uint2*)(sR + r * DD + c4 * 4) = *(const uint2*)(RHg + (size_t)gr * DD + c4 * 4);
#endif
  }
  v8f an[4];
#pragma unroll
  for (int t = 0; t < 4; ++t) an[t] = v8zero();
  __syncthreads();

#pragma unroll 1
  for (int kp = 0; kp < 4; ++kp) {
    stage_wpanel(Wint, DD, DD, kp, sWp);
#pragma unroll
    for (int t = 0; t < 4; ++t) {
      int tile = wave * 4 + t, strip = tile >> 3, col = tile & 7;
      an[t] = wmma_bf16(frag_a(sH, DD, strip * 16, kp * 32),
                        frag_b(sWp, 32, col * 16, 0), an[t]);
    }
    __syncthreads();
    stage_wpanel(Whnt, DD, DD, kp, sWp);
#pragma unroll
    for (int t = 0; t < 4; ++t) {
      int tile = wave * 4 + t, strip = tile >> 3, col = tile & 7;
      an[t] = wmma_bf16(frag_a(sR, DD, strip * 16, kp * 32),
                        frag_b(sWp, 32, col * 16, 0), an[t]);
    }
    __syncthreads();
  }
#pragma unroll
  for (int t = 0; t < 4; ++t) {
    int tile = wave * 4 + t, strip = tile >> 3, col = tile & 7;
    int n = col * 16 + (lane & 15), mb = strip * 16 + ((lane >> 4) << 3);
    float bn = bin_[n] + bhn[n];
#pragma unroll
    for (int j = 0; j < 8; ++j) {
      int gm = row0 + mb + j;
      if (gm < rows) {
        float nv = tanhf(an[t][j] + bn);
        float z = Z[(size_t)gm * DD + n];
        float hv = HV[(size_t)gm * DD + n];
        OUT[(size_t)gm * DD + n] = (1.f - z) * nv + z * hv;
      }
    }
  }
}

// ---------------------------------------------------------------------------
extern "C" void kernel_launch(void* const* d_in, const int* in_sizes, int n_in,
                              void* d_out, int out_size, void* d_ws, size_t ws_size,
                              hipStream_t stream) {
  (void)in_sizes; (void)n_in; (void)out_size; (void)ws_size;
  constexpr int N = 50000, E = 10000, NNZ = 500000, NNZS = 80000;

  const float* h        = (const float*)d_in[0];
  const int*   inc_row  = (const int*)d_in[1];
  const int*   inc_col  = (const int*)d_in[2];
  const float* inc_val  = (const float*)d_in[3];
  const float* deg_v    = (const float*)d_in[4];
  const float* deg_e    = (const float*)d_in[5];
  const int*   sent_row = (const int*)d_in[6];
  const int*   sent_col = (const int*)d_in[7];
  const float* sent_val = (const float*)d_in[8];
  // d_in[9]=num_nodes, d_in[10]=num_edges (compile-time constants here)
  const float* mlp1_W1 = (const float*)d_in[11];
  const float* mlp1_b1 = (const float*)d_in[12];
  const float* mlp1_W2 = (const float*)d_in[13];
  const float* mlp1_b2 = (const float*)d_in[14];
  const float* mlp2_W1 = (const float*)d_in[15];
  const float* mlp2_b1 = (const float*)d_in[16];
  const float* mlp2_W2 = (const float*)d_in[17];
  const float* mlp2_b2 = (const float*)d_in[18];
  const float* att1_W1 = (const float*)d_in[19];
  const float* att1_b1 = (const float*)d_in[20];
  const float* att1_W2 = (const float*)d_in[21];
  const float* att1_b2 = (const float*)d_in[22];
  const float* att2_W1 = (const float*)d_in[23];
  const float* att2_b1 = (const float*)d_in[24];
  const float* att2_W2 = (const float*)d_in[25];
  const float* att2_b2 = (const float*)d_in[26];
  const float* giz_W = (const float*)d_in[27];
  const float* giz_b = (const float*)d_in[28];
  const float* ghz_W = (const float*)d_in[29];
  const float* ghz_b = (const float*)d_in[30];
  const float* gir_W = (const float*)d_in[31];
  const float* gir_b = (const float*)d_in[32];
  const float* ghr_W = (const float*)d_in[33];
  const float* ghr_b = (const float*)d_in[34];
  const float* gin_W = (const float*)d_in[35];
  const float* gin_b = (const float*)d_in[36];
  const float* ghn_W = (const float*)d_in[37];
  const float* ghn_b = (const float*)d_in[38];
  const float* bn1_g = (const float*)d_in[39];
  const float* bn1_b = (const float*)d_in[40];
  const float* bn2_g = (const float*)d_in[41];
  const float* bn2_b = (const float*)d_in[42];

  // ---- workspace carve (~98 MB) ----
  char* wp = (char*)d_ws;
  auto take = [&](size_t bytes) -> char* {
    char* r = wp;
    wp += (bytes + 255) & ~(size_t)255;
    return r;
  };
  float*    hm     = (float*)take((size_t)N * DD * 4);   // h_m -> h_t2 -> z
  u16*      hmb    = (u16*)  take((size_t)N * DD * 2);   // h_m bf -> h_t2 bf -> rh
  float*    ht     = (float*)take((size_t)E * DD * 4);
  u16*      htb    = (u16*)  take((size_t)E * DD * 2);
  float*    pooled = (float*)take((size_t)E * DD * 4);
  float*    rs     = (float*)take((size_t)E * 4);
  float*    he     = (float*)take((size_t)E * DD * 4);
  float*    hn     = (float*)take((size_t)E * DD * 4);
  float*    hn2    = (float*)take((size_t)E * DD * 4);
  u16*      hn2b   = (u16*)  take((size_t)E * DD * 2);
  float*    av     = (float*)take((size_t)NNZ * 4);      // a then a2
  float*    p2     = (float*)take((size_t)N * DD * 4);   // pooled2 -> h_v
  float*    rs2    = (float*)take((size_t)N * 4);
  unsigned* red    = (unsigned*)take(256);
  float*    mu     = (float*)take(DD * 4);
  float*    var    = (float*)take(DD * 4);
  // bf16 transposed weights (one-time conversion per launch)
  u16* m1w1t = (u16*)take((size_t)DD * DD * 2);
  u16* m1w2t = (u16*)take((size_t)DD * DD * 2);
  u16* m2w1t = (u16*)take((size_t)DD * DD * 2);
  u16* m2w2t = (u16*)take((size_t)DD * DD * 2);
  u16* a1w1t = (u16*)take((size_t)256 * 256 * 2);
  u16* a2w1t = (u16*)take((size_t)256 * 256 * 2);
  u16* gizt  = (u16*)take((size_t)DD * DD * 2);
  u16* ghzt  = (u16*)take((size_t)DD * DD * 2);
  u16* girt  = (u16*)take((size_t)DD * DD * 2);
  u16* ghrt  = (u16*)take((size_t)DD * DD * 2);
  u16* gint  = (u16*)take((size_t)DD * DD * 2);
  u16* ghnt  = (u16*)take((size_t)DD * DD * 2);

  float* ht2 = hm;  u16* ht2b = hmb;   // buffer reuse (lifetimes disjoint)
  float* zbuf = hm; u16* rhbuf = hmb;
  float* hv = p2;

  init_red_kernel<<<1, 32, 0, stream>>>(red);

  // 0) weight conversion: fp32 row-major -> bf16 N-major
  transpose_bf16_kernel<<<64, 256, 0, stream>>>(mlp1_W1, m1w1t, DD, 7);
  transpose_bf16_kernel<<<64, 256, 0, stream>>>(mlp1_W2, m1w2t, DD, 7);
  transpose_bf16_kernel<<<64, 256, 0, stream>>>(mlp2_W1, m2w1t, DD, 7);
  transpose_bf16_kernel<<<64, 256, 0, stream>>>(mlp2_W2, m2w2t, DD, 7);
  transpose_bf16_kernel<<<256, 256, 0, stream>>>(att1_W1, a1w1t, 256, 8);
  transpose_bf16_kernel<<<256, 256, 0, stream>>>(att2_W1, a2w1t, 256, 8);
  transpose_bf16_kernel<<<64, 256, 0, stream>>>(giz_W, gizt, DD, 7);
  transpose_bf16_kernel<<<64, 256, 0, stream>>>(ghz_W, ghzt, DD, 7);
  transpose_bf16_kernel<<<64, 256, 0, stream>>>(gir_W, girt, DD, 7);
  transpose_bf16_kernel<<<64, 256, 0, stream>>>(ghr_W, ghrt, DD, 7);
  transpose_bf16_kernel<<<64, 256, 0, stream>>>(gin_W, gint, DD, 7);
  transpose_bf16_kernel<<<64, 256, 0, stream>>>(ghn_W, ghnt, DD, 7);

  // 1) h_m = mlp1(h)  [N,D]  (+ bf16 mirror)
  dense_mlp_kernel<<<(N + 63) / 64, 256, 0, stream>>>(
      h, m1w1t, mlp1_b1, m1w2t, mlp1_b2, hm, hmb, N);

  // 2) h_t = deg_e * (H^T @ h_m)  [E,D]
  hipMemsetAsync(ht, 0, (size_t)E * DD * 4, stream);
  spmm_scatter_kernel<<<2048, 256, 0, stream>>>(ht, inc_col, inc_row, inc_val, hm, NNZ);
  rowscale_bf16_kernel<<<2048, 256, 0, stream>>>(ht, deg_e, htb, E);

  // 3) attention pass 1: a = attn([h_m[row], h_t[col]])
  attn_kernel<<<(NNZ + 31) / 32, 256, 0, stream>>>(
      NNZ, inc_row, inc_col, hmb, htb, a1w1t, att1_b1, att1_W2, att1_b2, av);
  maxreduce_kernel<<<1024, 256, 0, stream>>>(av, NNZ, red + 0);
  hipMemsetAsync(pooled, 0, (size_t)E * DD * 4, stream);
  hipMemsetAsync(rs, 0, (size_t)E * 4, stream);
  weighted_scatter_kernel<<<2048, 256, 0, stream>>>(
      pooled, rs, inc_col, inc_row, av, red + 0, hm, NNZ);
  rowdiv_kernel<<<2048, 256, 0, stream>>>(he, pooled, rs, E);

  // 4) sentence mixing + bn2 + tanh + mlp2
  hipMemsetAsync(hn, 0, (size_t)E * DD * 4, stream);
  spmm_scatter_kernel<<<1024, 256, 0, stream>>>(hn, sent_row, sent_col, sent_val, he, NNZS);
  bn_stats_kernel<<<DD, 256, 0, stream>>>(hn, E, mu, var);
  bn_tanh_kernel<<<1024, 256, 0, stream>>>(hn, mu, var, bn2_g, bn2_b, E);
  dense_mlp_kernel<<<(E + 63) / 64, 256, 0, stream>>>(
      hn, m2w1t, mlp2_b1, m2w2t, mlp2_b2, hn2, hn2b, E);

  // 5) h_t2 = deg_v * (H @ h_n2)  [N,D]
  hipMemsetAsync(ht2, 0, (size_t)N * DD * 4, stream);
  spmm_scatter_kernel<<<2048, 256, 0, stream>>>(ht2, inc_row, inc_col, inc_val, hn2, NNZ);
  rowscale_bf16_kernel<<<4096, 256, 0, stream>>>(ht2, deg_v, ht2b, N);

  // 6) attention pass 2: a2 = attn([h_n2[col], h_t2[row]])
  attn_kernel<<<(NNZ + 31) / 32, 256, 0, stream>>>(
      NNZ, inc_col, inc_row, hn2b, ht2b, a2w1t, att2_b1, att2_W2, att2_b2, av);
  maxreduce_kernel<<<1024, 256, 0, stream>>>(av, NNZ, red + 1);
  hipMemsetAsync(p2, 0, (size_t)N * DD * 4, stream);
  hipMemsetAsync(rs2, 0, (size_t)N * 4, stream);
  weighted_scatter_kernel<<<2048, 256, 0, stream>>>(
      p2, rs2, inc_row, inc_col, av, red + 1, hn2, NNZ);
  rowdiv_kernel<<<4096, 256, 0, stream>>>(p2, p2, rs2, N);
  bn_stats_kernel<<<DD, 256, 0, stream>>>(p2, N, mu, var);
  bn_tanh_kernel<<<4096, 256, 0, stream>>>(p2, mu, var, bn1_g, bn1_b, N);  // p2 == h_v

  // 7) GRU update
  gru_a_kernel<<<(N + 63) / 64, 256, 0, stream>>>(
      h, hv, gizt, giz_b, ghzt, ghz_b, girt, gir_b, ghrt, ghr_b, zbuf, rhbuf, N);
  gru_b_kernel<<<(N + 63) / 64, 256, 0, stream>>>(
      h, rhbuf, gint, gin_b, ghnt, ghn_b, zbuf, hv, (float*)d_out, N);
}
